// NodeModel_23673859735571
// MI455X (gfx1250) — matrix operand order repaired and checked by reading, not compiled, not run on hardware
//
#include <hip/hip_runtime.h>

#define FN  128   // node feature dim
#define FE  64    // edge feature dim
#define E1  192   // FN+FE   (edge MLP input)
#define E2  256   // FN+L    (node MLP input)
#define LCH 128   // L (edge MLP output channels)
#define EPSV 1e-5f

typedef __attribute__((ext_vector_type(16))) __bf16 v16bf;
typedef __attribute__((ext_vector_type(8)))  float  v8f;

static __device__ __forceinline__ unsigned short f2bf(float f){
  union { float f; unsigned u; } v; v.f = f;
  unsigned u = v.u;
  return (unsigned short)((u + 0x7FFFu + ((u >> 16) & 1u)) >> 16); // RNE
}
static __device__ __forceinline__ float bf2f(unsigned short h){
  union { unsigned u; float f; } v; v.u = ((unsigned)h) << 16;
  return v.f;
}

// ---------------- elementwise f32 -> bf16 ----------------
__global__ void k_convert(const float* __restrict__ in, unsigned short* __restrict__ out, long long n){
  long long i  = (long long)blockIdx.x * blockDim.x + threadIdx.x;
  long long st = (long long)gridDim.x * blockDim.x;
  for (; i < n; i += st) out[i] = f2bf(in[i]);
}

// ---- column slice [ncol0, ncol0+Ncols) of W (K x Nfull, f32 row-major) ->
// ---- WMMA B-fragment order, bf16.
// frag(kt,nt): 32 lanes x 16 bf16. lane l (h=l>>4, n=ncol0+nt*16+(l&15)), elem j -> K = kt*32 + (j>>3)*16 + h*8 + (j&7)
__global__ void k_permuteB(const float* __restrict__ W, unsigned short* __restrict__ Wp,
                           int K, int Nfull, int ncol0, int Ncols){
  int Ntiles = Ncols >> 4;
  long long tot = (long long)K * Ncols;
  long long i  = (long long)blockIdx.x * blockDim.x + threadIdx.x;
  long long st = (long long)gridDim.x * blockDim.x;
  for (; i < tot; i += st){
    int frag = (int)(i >> 9);
    int rem  = (int)(i & 511);
    int lane = rem >> 4, j = rem & 15;
    int kt = frag / Ntiles, nt = frag % Ntiles;
    int k = kt*32 + (j >> 3)*16 + (lane >> 4)*8 + (j & 7);
    int n = ncol0 + nt*16 + (lane & 15);
    Wp[i] = f2bf(W[(long long)k * Nfull + n]);
  }
}

// ---------------- edge GEMM: [E,192]x[192,128] with BN stats ----------------
// kt loops kept rolled (split at the xh/eah boundary) so only one kt-slice of
// B (64 VGPRs) is live at a time: no full-W register hoist, no scratch spills,
// high occupancy to hide the random x[row] gather latency.
__global__ __launch_bounds__(256)
void k_gemm_edge(const unsigned short* __restrict__ xh,   // [N,128] bf16 (L2-resident, 12.8MB)
                 const unsigned short* __restrict__ eah,  // [E,64]  bf16
                 const int*            __restrict__ erow, // gather indices
                 const unsigned short* __restrict__ W1p,  // frag-ordered, 48KB
                 unsigned short*       __restrict__ Z1,   // [E,128] bf16 out (pre-BN)
                 float* __restrict__ gsum, float* __restrict__ gsq,
                 int E)
{
  __shared__ unsigned short sW[E1*LCH];  // 48KB
  __shared__ float sSum[LCH], sSq[LCH];
  int tid = threadIdx.x;
  { // cooperative 128b copy of W1 fragments into LDS
    const uint4* src = (const uint4*)W1p;
    uint4* dst = (uint4*)sW;
    for (int i = tid; i < (E1*LCH)/8; i += 256) dst[i] = src[i];
  }
  if (tid < LCH){ sSum[tid] = 0.f; sSq[tid] = 0.f; }
  __syncthreads();

  int wid = tid >> 5, lane = tid & 31;
  int m = lane & 15, hh = lane >> 4;
  int tiles = E >> 4;
  for (int tile = blockIdx.x*8 + wid; tile < tiles; tile += gridDim.x*8){
    int e = tile*16 + m;
    const unsigned short* xr = xh + (size_t)erow[e]*FN;
    const unsigned short* er = eah + (size_t)e*FE;
    v8f acc[8] = {};
    // kt = 0..3 : A from gathered x row (features 0..127)
#pragma unroll 1
    for (int kt = 0; kt < 4; ++kt){
      v16bf a;
      ((uint4*)&a)[0] = *(const uint4*)(xr + kt*32 + hh*8);
      ((uint4*)&a)[1] = *(const uint4*)(xr + kt*32 + 16 + hh*8);
      const unsigned short* wk = sW + ((kt*8)*32 + lane)*16;
#pragma unroll
      for (int nt = 0; nt < 8; ++nt){
        v16bf b;
        const uint4* bp = (const uint4*)(wk + nt*32*16);
        ((uint4*)&b)[0] = bp[0];
        ((uint4*)&b)[1] = bp[1];
        acc[nt] = __builtin_amdgcn_wmma_f32_16x16x32_bf16(false, a, false, b,
                                                          (short)0, acc[nt], false, false);
      }
    }
    // kt = 4..5 : A from edge_attr row (features 128..191)
#pragma unroll 1
    for (int kt2 = 0; kt2 < 2; ++kt2){
      v16bf a;
      ((uint4*)&a)[0] = *(const uint4*)(er + kt2*32 + hh*8);
      ((uint4*)&a)[1] = *(const uint4*)(er + kt2*32 + 16 + hh*8);
      const unsigned short* wk = sW + (((kt2 + 4)*8)*32 + lane)*16;
#pragma unroll
      for (int nt = 0; nt < 8; ++nt){
        v16bf b;
        const uint4* bp = (const uint4*)(wk + nt*32*16);
        ((uint4*)&b)[0] = bp[0];
        ((uint4*)&b)[1] = bp[1];
        acc[nt] = __builtin_amdgcn_wmma_f32_16x16x32_bf16(false, a, false, b,
                                                          (short)0, acc[nt], false, false);
      }
    }
    // store pre-BN Z (bf16) + accumulate per-channel sum / sumsq
#pragma unroll
    for (int nt = 0; nt < 8; ++nt){
      int n = nt*16 + m;
      float s = 0.f, q = 0.f;
#pragma unroll
      for (int r = 0; r < 8; ++r){
        float z = acc[nt][r];                 // C layout: row M = r + 8*hh, col n
        s += z; q += z*z;
        Z1[(size_t)(tile*16 + r + 8*hh)*LCH + n] = f2bf(z);
      }
      atomicAdd(&sSum[n], s);
      atomicAdd(&sSq[n],  q);
    }
  }
  __syncthreads();
  if (tid < LCH){ atomicAdd(&gsum[tid], sSum[tid]); atomicAdd(&gsq[tid], sSq[tid]); }
}

// ---------------- BN finalize: scale/shift (bias cancels in train-mode BN) ----------------
__global__ void k_finalize(const float* __restrict__ sum, const float* __restrict__ sq,
                           const float* __restrict__ gamma, const float* __restrict__ beta,
                           float invCnt, float* __restrict__ scale, float* __restrict__ shift, int C){
  int c = blockIdx.x*blockDim.x + threadIdx.x;
  if (c < C){
    float mu  = sum[c]*invCnt;
    float var = sq[c]*invCnt - mu*mu;
    float sc  = gamma[c]*rsqrtf(var + EPSV);
    scale[c] = sc;
    shift[c] = beta[c] - mu*sc;
  }
}

// ---------------- scatter-add (affine+ReLU fused) ----------------
__global__ __launch_bounds__(256)
void k_scatter(const unsigned short* __restrict__ Z1, const int* __restrict__ ecol,
               const float* __restrict__ scale, const float* __restrict__ shift,
               float* __restrict__ sacc, float* __restrict__ cnt, int E)
{
  long long t = (long long)blockIdx.x*blockDim.x + threadIdx.x;
  if (t >= (long long)E*32) return;
  int e = (int)(t >> 5);
  int lane = (int)(t & 31);
  int c0 = lane*4;
  int col = ecol[e];
  uint2 raw = *(const uint2*)(Z1 + (size_t)e*LCH + c0);
  unsigned short rr[4] = { (unsigned short)(raw.x & 0xffff), (unsigned short)(raw.x >> 16),
                           (unsigned short)(raw.y & 0xffff), (unsigned short)(raw.y >> 16) };
#pragma unroll
  for (int i = 0; i < 4; ++i){
    float z = fmaxf(bf2f(rr[i])*scale[c0+i] + shift[c0+i], 0.f);
    atomicAdd(&sacc[(size_t)col*LCH + c0 + i], z);
  }
  if (lane == 0) atomicAdd(&cnt[col], 1.f);
}

// ---------------- h2 = [x | mean-agg]  -> bf16 ----------------
__global__ void k_build_h2(const float* __restrict__ x, const float* __restrict__ sacc,
                           const float* __restrict__ cnt, unsigned short* __restrict__ h2h, int N){
  long long t  = (long long)blockIdx.x*blockDim.x + threadIdx.x;
  if (t >= (long long)N*E2) return;
  int node = (int)(t >> 8);
  int c    = (int)(t & 255);
  float v;
  if (c < FN) v = x[(size_t)node*FN + c];
  else {
    float k = cnt[node];
    v = (k > 0.f) ? sacc[(size_t)node*LCH + (c - FN)] / k : 0.f;
  }
  h2h[t] = f2bf(v);
}

// ------- node GEMM half: [N,256] x [256,128] (column slice) with BN stats -------
// kt loop kept rolled: only one kt-slice of B (64 VGPRs) live at a time ->
// no full-W register hoist, no scratch spills. W half (64KB) stays L0/L2-resident.
__global__ __launch_bounds__(256)
void k_gemm_node8(const unsigned short* __restrict__ A,   // [N,256] bf16
                  const unsigned short* __restrict__ Wp,  // frag-ordered 256x128 half, 64KB
                  unsigned short* __restrict__ Z,         // [N,256] bf16 out (pre-BN)
                  float* __restrict__ gsum, float* __restrict__ gsq,
                  int N, int ncol0)
{
  __shared__ float sSum[LCH], sSq[LCH];
  int tid = threadIdx.x;
  if (tid < LCH){ sSum[tid] = 0.f; sSq[tid] = 0.f; }
  __syncthreads();

  int wid = tid >> 5, lane = tid & 31;
  int m = lane & 15, hh = lane >> 4;
  int tiles = N >> 4;
  for (int tile = blockIdx.x*8 + wid; tile < tiles; tile += gridDim.x*8){
    const unsigned short* ar = A + (size_t)(tile*16 + m)*E2;
    v8f acc[8] = {};
#pragma unroll 1
    for (int kt = 0; kt < 8; ++kt){
      v16bf a;
      ((uint4*)&a)[0] = *(const uint4*)(ar + kt*32 + hh*8);
      ((uint4*)&a)[1] = *(const uint4*)(ar + kt*32 + 16 + hh*8);
      const unsigned short* wk = Wp + ((size_t)kt*8*32 + lane)*16;
#pragma unroll
      for (int nt = 0; nt < 8; ++nt){
        v16bf b;
        const uint4* bp = (const uint4*)(wk + nt*32*16);
        ((uint4*)&b)[0] = bp[0];
        ((uint4*)&b)[1] = bp[1];
        acc[nt] = __builtin_amdgcn_wmma_f32_16x16x32_bf16(false, a, false, b,
                                                          (short)0, acc[nt], false, false);
      }
    }
#pragma unroll
    for (int nt = 0; nt < 8; ++nt){
      int n = nt*16 + m;
      float s = 0.f, q = 0.f;
#pragma unroll
      for (int r = 0; r < 8; ++r){
        float z = acc[nt][r];
        s += z; q += z*z;
        Z[(size_t)(tile*16 + r + 8*hh)*E2 + ncol0 + n] = f2bf(z);
      }
      atomicAdd(&sSum[n], s);
      atomicAdd(&sSq[n],  q);
    }
  }
  __syncthreads();
  if (tid < LCH){ atomicAdd(&gsum[ncol0 + tid], sSum[tid]); atomicAdd(&gsq[ncol0 + tid], sSq[tid]); }
}

// ---------------- elementwise BN2 affine + ReLU -> bf16 ----------------
__global__ void k_bn_relu(const unsigned short* __restrict__ Z, const float* __restrict__ scale,
                          const float* __restrict__ shift, unsigned short* __restrict__ H, long long n){
  long long t = (long long)blockIdx.x*blockDim.x + threadIdx.x;
  if (t >= n) return;
  int c = (int)(t & 255);
  H[t] = f2bf(fmaxf(bf2f(Z[t])*scale[c] + shift[c], 0.f));
}

// ---------------- final GEMM: [N,256]x[256,128] + bias + ReLU -> f32 out ----------------
__global__ __launch_bounds__(256)
void k_gemm_final(const unsigned short* __restrict__ A, const unsigned short* __restrict__ Wp,
                  const float* __restrict__ bias, float* __restrict__ out, int N)
{
  int tid = threadIdx.x;
  int wid = tid >> 5, lane = tid & 31;
  int m = lane & 15, hh = lane >> 4;
  int tiles = N >> 4;
  for (int tile = blockIdx.x*8 + wid; tile < tiles; tile += gridDim.x*8){
    const unsigned short* ar = A + (size_t)(tile*16 + m)*E2;
    v8f acc[8] = {};
#pragma unroll 1
    for (int kt = 0; kt < 8; ++kt){
      v16bf a;
      ((uint4*)&a)[0] = *(const uint4*)(ar + kt*32 + hh*8);
      ((uint4*)&a)[1] = *(const uint4*)(ar + kt*32 + 16 + hh*8);
      const unsigned short* wk = Wp + ((size_t)kt*8*32 + lane)*16;
#pragma unroll
      for (int nt = 0; nt < 8; ++nt){
        v16bf b;
        const uint4* bp = (const uint4*)(wk + nt*32*16);
        ((uint4*)&b)[0] = bp[0];
        ((uint4*)&b)[1] = bp[1];
        acc[nt] = __builtin_amdgcn_wmma_f32_16x16x32_bf16(false, a, false, b,
                                                          (short)0, acc[nt], false, false);
      }
    }
#pragma unroll
    for (int nt = 0; nt < 8; ++nt){
      int n = nt*16 + m;
      float bv = bias[n];
#pragma unroll
      for (int r = 0; r < 8; ++r)
        out[(size_t)(tile*16 + r + 8*hh)*FN + n] = fmaxf(acc[nt][r] + bv, 0.f);
    }
  }
}

extern "C" void kernel_launch(void* const* d_in, const int* in_sizes, int n_in,
                              void* d_out, int out_size, void* d_ws, size_t ws_size,
                              hipStream_t stream)
{
  (void)n_in; (void)out_size; (void)ws_size;
  const float* x   = (const float*)d_in[0];
  const int*   eix = (const int*)d_in[1];
  const float* ea  = (const float*)d_in[2];
  // d_in[3]=u, d_in[4]=batch unused; b1 (d_in[6]) and b2 (d_in[10]) cancel inside BN
  const float* W1  = (const float*)d_in[5];
  const float* g1  = (const float*)d_in[7];
  const float* be1 = (const float*)d_in[8];
  const float* W2  = (const float*)d_in[9];
  const float* g2  = (const float*)d_in[11];
  const float* be2 = (const float*)d_in[12];
  const float* Wl  = (const float*)d_in[13];
  const float* bl  = (const float*)d_in[14];
  float* out = (float*)d_out;

  const int N = in_sizes[0] / FN;
  const int E = in_sizes[1] / 2;
  const int* erow = eix;
  const int* ecol = eix + E;

  // workspace carve-out (256B aligned slabs)
  char* w = (char*)d_ws;
  auto take = [&](size_t bytes) -> char* {
    char* p = w; w += (bytes + 255) & ~(size_t)255; return p;
  };
  unsigned short* xh   = (unsigned short*)take((size_t)N * FN  * 2);
  unsigned short* eah  = (unsigned short*)take((size_t)E * FE  * 2);
  unsigned short* W1p  = (unsigned short*)take((size_t)E1 * LCH * 2);
  unsigned short* W2pA = (unsigned short*)take((size_t)E2 * LCH * 2);  // cols   0..127
  unsigned short* W2pB = (unsigned short*)take((size_t)E2 * LCH * 2);  // cols 128..255
  unsigned short* Wlp  = (unsigned short*)take((size_t)E2 * FN  * 2);
  unsigned short* Z1   = (unsigned short*)take((size_t)E * LCH * 2);
  unsigned short* Z2   = (unsigned short*)take((size_t)N * E2  * 2);
  unsigned short* H2   = (unsigned short*)take((size_t)N * E2  * 2);
  unsigned short* h2h  = (unsigned short*)take((size_t)N * E2  * 2);
  float* sacc = (float*)take((size_t)N * LCH * 4);
  float* cnt  = (float*)take((size_t)N * 4);
  float* st1  = (float*)take(2 * LCH * 4);  // sum | sumsq
  float* sc1  = (float*)take(2 * LCH * 4);  // scale | shift
  float* st2  = (float*)take(2 * E2 * 4);
  float* sc2  = (float*)take(2 * E2 * 4);

  hipMemsetAsync(sacc, 0, (size_t)N * LCH * 4, stream);
  hipMemsetAsync(cnt,  0, (size_t)N * 4, stream);
  hipMemsetAsync(st1,  0, 2 * LCH * 4, stream);
  hipMemsetAsync(st2,  0, 2 * E2  * 4, stream);

  // precision/layout conversions
  long long nx = (long long)N * FN;
  k_convert<<<(int)((nx + 255) / 256), 256, 0, stream>>>(x, xh, nx);
  long long ne = (long long)E * FE;
  k_convert<<<8192, 256, 0, stream>>>(ea, eah, ne);
  k_permuteB<<<(E1*LCH + 255)/256, 256, 0, stream>>>(W1, W1p,  E1, LCH, 0,   LCH);
  k_permuteB<<<(E2*LCH + 255)/256, 256, 0, stream>>>(W2, W2pA, E2, E2,  0,   LCH);
  k_permuteB<<<(E2*LCH + 255)/256, 256, 0, stream>>>(W2, W2pB, E2, E2,  LCH, LCH);
  k_permuteB<<<(E2*FN  + 255)/256, 256, 0, stream>>>(Wl, Wlp,  E2, FN,  0,   FN);

  // stage 1: edge MLP (WMMA) + BN1 stats
  int etiles = E / 16;
  k_gemm_edge<<<(etiles + 7)/8, 256, 0, stream>>>(xh, eah, erow, W1p, Z1, st1, st1 + LCH, E);
  k_finalize<<<1, LCH, 0, stream>>>(st1, st1 + LCH, g1, be1, 1.f/(float)E, sc1, sc1 + LCH, LCH);

  // stage 2: fused BN1-affine + ReLU + scatter-add
  long long sth = (long long)E * 32;
  k_scatter<<<(int)((sth + 255)/256), 256, 0, stream>>>(Z1, ecol, sc1, sc1 + LCH, sacc, cnt, E);

  // stage 3: h2 = [x | mean-agg], node MLP (WMMA, two column halves) + BN2 stats
  long long nh2 = (long long)N * E2;
  k_build_h2<<<(int)((nh2 + 255)/256), 256, 0, stream>>>(x, sacc, cnt, h2h, N);
  int ntiles = N / 16;
  k_gemm_node8<<<(ntiles + 7)/8, 256, 0, stream>>>(h2h, W2pA, Z2, st2, st2 + E2, N, 0);
  k_gemm_node8<<<(ntiles + 7)/8, 256, 0, stream>>>(h2h, W2pB, Z2, st2, st2 + E2, N, LCH);
  k_finalize<<<1, E2, 0, stream>>>(st2, st2 + E2, g2, be2, 1.f/(float)N, sc2, sc2 + E2, E2);
  k_bn_relu<<<(int)((nh2 + 255)/256), 256, 0, stream>>>(Z2, sc2, sc2 + E2, H2, nh2);

  // stage 4: final linear (WMMA) + bias + ReLU -> f32 output
  k_gemm_final<<<(ntiles + 7)/8, 256, 0, stream>>>(H2, Wlp, bl, out, N);
}
